// SSM_26963804684669
// MI455X (gfx1250) — compile-verified, matrix operand order checked by reading, not run on hardware
//
#include <hip/hip_runtime.h>

#define DEV __device__ __forceinline__

typedef __bf16 bf16;
typedef __attribute__((ext_vector_type(16))) __bf16 v16bf;
typedef __attribute__((ext_vector_type(8)))  __bf16 v8bf;
typedef __attribute__((ext_vector_type(8)))  float  v8f;
typedef __attribute__((ext_vector_type(4)))  float  f4;
typedef __attribute__((ext_vector_type(4)))  int    v4i;

// ---- CDNA5 async global->LDS copy path (guarded; sync LDS fallback otherwise) ----
#if defined(__has_builtin)
#if __has_builtin(__builtin_amdgcn_global_load_async_to_lds_b128) && \
    __has_builtin(__builtin_amdgcn_s_wait_asynccnt)
#define ASYNC_COPY 1
#endif
#endif

#ifdef ASYNC_COPY
// builtin signature (from hipcc diagnostic): (int4 AS1* src, int4 AS3* dst, imm offset, imm cpol)
typedef __attribute__((address_space(1))) v4i* g_v4i;
typedef __attribute__((address_space(3))) v4i* l_v4i;
#endif

// ---- problem dims ----
constexpr int DMODEL  = 256;
constexpr int DSTATE  = 128;
constexpr int DINNER  = 512;
constexpr int CHUNK   = 64;
constexpr int CONVDIM = DINNER + 2*DSTATE;      // 768
constexpr int NZXC    = DINNER + CONVDIM;       // 1280 used cols of in_proj
constexpr int BATCH   = 8;
constexpr int SEQ     = 8192;
constexpr int ROWS    = BATCH*SEQ;              // 65536
constexpr int TCH     = SEQ/CHUNK;              // 128 chunks / batch
constexpr int NBT     = BATCH*TCH;              // 1024 chunks

// ---- workspace layout (bytes, all 256-aligned) ----
constexpr size_t OFF_W1B   = 0;                                   // 1280*256 bf16
constexpr size_t OFF_W2B   = OFF_W1B + (size_t)NZXC*DMODEL*2;     // 256*512 bf16
constexpr size_t OFF_SPZ   = OFF_W2B + (size_t)DMODEL*DINNER*2;   // softplus(z) f32 [ROWS][512]
constexpr size_t OFF_XT    = OFF_SPZ + (size_t)ROWS*DINNER*4;     // x chunk-transposed bf16 [NBT][512][64]
constexpr size_t OFF_BBF   = OFF_XT  + (size_t)ROWS*DINNER*2;     // B natural bf16 [ROWS][128]
constexpr size_t OFF_BDT   = OFF_BBF + (size_t)ROWS*DSTATE*2;     // B*decay chunk-transposed bf16 [NBT][128][64]
constexpr size_t OFF_CBF   = OFF_BDT + (size_t)ROWS*DSTATE*2;     // C natural bf16 [ROWS][128]
constexpr size_t OFF_YBF   = OFF_CBF + (size_t)ROWS*DSTATE*2;     // gated y bf16 [ROWS][512]
constexpr size_t OFF_UNION = OFF_YBF + (size_t)ROWS*DINNER*2;     // xBC f32 (201MB) then PT f32 (268MB)
// PT: prefix states f32 [NBT][512][128]  (aliases xBC: xBC dead before PT written)

DEV float softplusf(float x){ return x > 20.f ? x : log1pf(expf(x)); }

DEV v8f wmma_bf(v16bf a, v16bf b, v8f c){
  return __builtin_amdgcn_wmma_f32_16x16x32_bf16(false, a, false, b, (short)0, c, false, false);
}

// Operand fragment for 16x16x32 bf16 WMMA.  Per ISA layout: lane = {r = lane&15, h = lane>>4};
// vector elems 0..7  <-> K = h*8 + 0..7 ;  elems 8..15 <-> K = 16 + h*8 + 0..7.
// `p` points at (row r, K=0) of the 32-wide K slice; all loads are contiguous 16B.
DEV v16bf ldfrag(const bf16* p, int h){
  v8bf lo = *(const v8bf*)(p + h*8);
  v8bf hi = *(const v8bf*)(p + 16 + h*8);
  return __builtin_shufflevector(lo, hi, 0,1,2,3,4,5,6,7,8,9,10,11,12,13,14,15);
}
DEV v16bf ldfrag_scale(const bf16* p, int h, float s){
  v8bf lo = *(const v8bf*)(p + h*8);
  v8bf hi = *(const v8bf*)(p + 16 + h*8);
  v16bf r;
#pragma unroll
  for(int i=0;i<8;i++){ r[i]=(__bf16)((float)lo[i]*s); r[8+i]=(__bf16)((float)hi[i]*s); }
  return r;
}
DEV v16bf ldfrag_f32(const float* p, int h){
  const float* q0 = p + h*8;
  const float* q1 = p + 16 + h*8;
  f4 a0=*(const f4*)q0, a1=*(const f4*)(q0+4);
  f4 b0=*(const f4*)q1, b1=*(const f4*)(q1+4);
  v16bf r;
#pragma unroll
  for(int i=0;i<4;i++){
    r[i]   =(__bf16)a0[i]; r[4+i] =(__bf16)a1[i];
    r[8+i] =(__bf16)b0[i]; r[12+i]=(__bf16)b1[i];
  }
  return r;
}

// Cooperative stage of a 128-row x 32-col bf16 weight tile (8KB) into LDS.
// 256 threads x 2 x 16B.  Async path uses ASYNCcnt-tracked global->LDS DMA.
DEV void stage_wtile(const bf16* __restrict__ W, int ldk, int n0, int k,
                     bf16* buf, int tid){
#pragma unroll
  for(int it=0; it<2; ++it){
    const int chunk = it*256 + tid;
    const int row = chunk >> 2, seg = chunk & 3;
    const bf16* src = W + (size_t)(n0 + row)*ldk + k + seg*8;
    bf16* dst = buf + row*32 + seg*8;
#ifdef ASYNC_COPY
    __builtin_amdgcn_global_load_async_to_lds_b128(
        (g_v4i)(const void*)src, (l_v4i)(void*)dst, 0, 0);
#else
    *(f4*)dst = *(const f4*)src;
#endif
  }
}
DEV void wait_stage(){
#ifdef ASYNC_COPY
  __builtin_amdgcn_s_wait_asynccnt(0);
#endif
}

// ---------------- K1: weight conversion to bf16 ----------------
__global__ __launch_bounds__(256) void k_cvt(const float* __restrict__ w1,
                                             const float* __restrict__ w2,
                                             bf16* __restrict__ w1b, bf16* __restrict__ w2b){
  int i = blockIdx.x*256 + threadIdx.x;
  if(i < NZXC*DMODEL)  w1b[i] = (__bf16)w1[i];
  if(i < DMODEL*DINNER) w2b[i] = (__bf16)w2[i];
}

// ---------------- K2: in_proj GEMM  zxc = u @ W1^T + b ----------------
// grid (ROWS/128, NZXC/128); wave tile 16M x 128N; 8 waves stacked along M.
// Weight tile double-buffered through LDS with async global->LDS copies.
__global__ __launch_bounds__(256) void k_gemm1(const float* __restrict__ u,
                                               const bf16* __restrict__ w1b,
                                               const float* __restrict__ bias,
                                               float* __restrict__ spz,
                                               float* __restrict__ xBC){
  __shared__ bf16 sb[2][128*32];
  const int tid = threadIdx.x;
  const int wave = tid >> 5, lane = tid & 31;
  const int h = lane >> 4, r = lane & 15;
  const int m0 = blockIdx.x*128 + wave*16;
  const int n0 = blockIdx.y*128;
  v8f acc[8] = {};
  const float* arow = u + (size_t)(m0 + r)*DMODEL;

  stage_wtile(w1b, DMODEL, n0, 0, sb[0], tid);
  int cur = 0;
#pragma unroll 1
  for(int k=0;k<DMODEL;k+=32){
    wait_stage();
    __syncthreads();
    if(k + 32 < DMODEL) stage_wtile(w1b, DMODEL, n0, k + 32, sb[cur^1], tid);
    if(k + 64 < DMODEL) __builtin_prefetch(arow + k + 64, 0, 0);
    const v16bf af = ldfrag_f32(arow + k, h);
#pragma unroll
    for(int j=0;j<8;j++)
      acc[j] = wmma_bf(af, ldfrag(&sb[cur][(j*16 + r)*32], h), acc[j]);
    __syncthreads();
    cur ^= 1;
  }
#pragma unroll
  for(int j=0;j<8;j++){
    const int n = n0 + j*16 + r;
    const float bv = bias[n];
#pragma unroll
    for(int i=0;i<8;i++){
      const size_t m = (size_t)m0 + i + 8*h;
      const float v = acc[j][i] + bv;
      if(n < DINNER) spz[m*DINNER + n] = softplusf(v);
      else           xBC[m*CONVDIM + (n - DINNER)] = v;
    }
  }
}

// ---------------- K3: causal dwconv(K=4) + softplus + splits / transposes ----------------
// grid (NBT, 4); block stages a 67x192 halo tile of xBC in LDS, writes x,Bd transposed per chunk.
__global__ __launch_bounds__(256) void k_conv(const float* __restrict__ xBC,
                                              const float* __restrict__ cw,
                                              const float* __restrict__ cb,
                                              const float* __restrict__ reward,
                                              const float* __restrict__ bpw,
                                              const float* __restrict__ bpb,
                                              const float* __restrict__ alog,
                                              bf16* __restrict__ xT,
                                              bf16* __restrict__ Bbf,
                                              bf16* __restrict__ BdT,
                                              bf16* __restrict__ Cbf){
  __shared__ float lds[67*192];
  const float a = -expf(alog[0]);
  const int bt = blockIdx.x, c0 = blockIdx.y*192;
  const int b = bt >> 7, t = bt & 127;
  const int tid = threadIdx.x;
  for(int idx = tid; idx < 67*192; idx += 256){
    const int rr = idx / 192, c = idx - rr*192;
    const int l = t*CHUNK + rr - 3;
    lds[idx] = (l >= 0) ? xBC[((size_t)b*SEQ + l)*CONVDIM + c0 + c] : 0.f;
  }
  __syncthreads();
  const int lpos = tid & 63;
  const int csub = tid >> 6;                       // 0..3
  const size_t row = (size_t)bt*CHUNK + lpos;      // global sequence row
#pragma unroll 1
  for(int cc = csub; cc < 192; cc += 4){
    const int cg = c0 + cc;
    float acc = cb[cg];
#pragma unroll
    for(int j=0;j<4;j++) acc += lds[(lpos + j)*192 + cc] * cw[cg*4 + j];
    const float act = softplusf(acc);
    if(cg < DINNER){
      xT[((size_t)bt*DINNER + cg)*CHUNK + lpos] = (__bf16)(-act);
    } else if(cg < DINNER + DSTATE){
      const int n = cg - DINNER;
      const float bm = act * (reward[row]*bpw[n] + bpb[n]);
      Bbf[row*DSTATE + n] = (__bf16)bm;
      BdT[((size_t)bt*DSTATE + n)*CHUNK + lpos] = (__bf16)(bm * expf(a*(float)(CHUNK-1-lpos)));
    } else {
      Cbf[row*DSTATE + (cg - DINNER - DSTATE)] = (__bf16)act;
    }
  }
}

// ---------------- K4: per-chunk states  CS = Bd^T(128x64) @ X(64x512) ----------------
// grid (NBT, 4); wave tile 16n x 128p; D stored transposed into PT[bt][p][n].
__global__ __launch_bounds__(256) void k_states(const bf16* __restrict__ BdT,
                                                const bf16* __restrict__ xT,
                                                float* __restrict__ PT){
  const int bt = blockIdx.x, p0 = blockIdx.y*128;
  const int wave = threadIdx.x >> 5, lane = threadIdx.x & 31;
  const int h = lane >> 4, r = lane & 15;
  const int n0 = wave*16;
  const bf16* arow = BdT + ((size_t)bt*DSTATE + n0 + r)*CHUNK;
  const v16bf a0 = ldfrag(arow, h);
  const v16bf a1 = ldfrag(arow + 32, h);
  v8f acc[8] = {};
#pragma unroll
  for(int j=0;j<8;j++){
    const bf16* brow = xT + ((size_t)bt*DINNER + p0 + j*16 + r)*CHUNK;
    acc[j] = wmma_bf(a0, ldfrag(brow,      h), acc[j]);
    acc[j] = wmma_bf(a1, ldfrag(brow + 32, h), acc[j]);
  }
#pragma unroll
  for(int j=0;j<8;j++){
    // lane holds column p = p0+j*16+r, rows n0+8h..n0+8h+7 -> contiguous 32B in PT[p][n]
    float* dst = PT + ((size_t)bt*DINNER + p0 + j*16 + r)*DSTATE + n0 + 8*h;
    *(v8f*)dst = acc[j];
  }
}

// ---------------- K5: inter-chunk scan  P[t] = lam*P[t-1] + CS[t-1], in place ----------------
__global__ __launch_bounds__(256) void k_scan(float* __restrict__ PT,
                                              const float* __restrict__ alog){
  const float a = -expf(alog[0]);
  const float lam = expf((float)CHUNK * a);
  const int idx = blockIdx.x*256 + threadIdx.x;        // over BATCH * (512*128/4)
  const int b = idx >> 14, r4 = idx & 16383;
  f4* base = (f4*)PT + ((size_t)b*TCH)*16384 + r4;
  f4 pref = {0.f,0.f,0.f,0.f};
#pragma unroll 1
  for(int t=0;t<TCH;t++){
    f4* p = base + (size_t)t*16384;
    f4 cs = *p;
    *p = pref;
    pref = pref*lam + cs;
  }
}

// ---------------- K6: Y = mask(C B^T * L) @ X  +  (C*exp(a(l+1))) @ P ; gate by softplus(z) ----------------
// grid (NBT, 2).  Phase A: 8 waves build G(64x64, K=128) and write decayed/masked M to LDS (bf16).
// Phase B: wave tile 16l x 128p, two K loops (M@X K=64, Cd@P K=128).
__global__ __launch_bounds__(256) void k_y(const bf16* __restrict__ Cbf,
                                           const bf16* __restrict__ Bbf,
                                           const bf16* __restrict__ xT,
                                           const float* __restrict__ PT,
                                           const float* __restrict__ spz,
                                           const float* __restrict__ alog,
                                           bf16* __restrict__ ybf){
  __shared__ bf16 Mlds[64*80];                 // row stride 80 keeps 16B alignment
  const float a = -expf(alog[0]);
  const int bt = blockIdx.x;
  const int wave = threadIdx.x >> 5, lane = threadIdx.x & 31;
  const int h = lane >> 4, r = lane & 15;
  const size_t row0 = (size_t)bt*CHUNK;        // global row base of this chunk

  { // ---- Phase A: G = C @ B^T ----
    const int lt = wave >> 1, st = wave & 1;   // 4 l-tiles x 2 s-halves
    const bf16* arow = Cbf + (row0 + lt*16 + r)*DSTATE;
    v8f g[2] = {};
#pragma unroll
    for(int k=0;k<DSTATE;k+=32){
      const v16bf af = ldfrag(arow + k, h);
#pragma unroll
      for(int j=0;j<2;j++){
        const bf16* brow = Bbf + (row0 + st*32 + j*16 + r)*DSTATE + k;
        g[j] = wmma_bf(af, ldfrag(brow, h), g[j]);
      }
    }
#pragma unroll
    for(int j=0;j<2;j++){
      const int s = st*32 + j*16 + r;
#pragma unroll
      for(int i=0;i<8;i++){
        const int l = lt*16 + i + 8*h;
        const float v = (l >= s) ? g[j][i]*expf(a*(float)(l - s)) : 0.f;
        Mlds[l*80 + s] = (__bf16)v;
      }
    }
  }
  __syncthreads();

  // ---- Phase B ----
  const int lt = wave >> 1;
  const int p0 = blockIdx.y*256 + (wave & 1)*128;
  v8f acc[8] = {};

  const bf16* mrow = &Mlds[(lt*16 + r)*80];
  const v16bf m0f = ldfrag(mrow,      h);
  const v16bf m1f = ldfrag(mrow + 32, h);
#pragma unroll
  for(int j=0;j<8;j++){
    const bf16* xrow = xT + ((size_t)bt*DINNER + p0 + j*16 + r)*CHUNK;
    acc[j] = wmma_bf(m0f, ldfrag(xrow,      h), acc[j]);
    acc[j] = wmma_bf(m1f, ldfrag(xrow + 32, h), acc[j]);
  }

  const float rscale = expf(a*(float)(lt*16 + r + 1));   // exp(a*(l+1)) folded into C row
  const bf16* crow = Cbf + (row0 + lt*16 + r)*DSTATE;
#pragma unroll
  for(int k=0;k<DSTATE;k+=32){
    const v16bf cf = ldfrag_scale(crow + k, h, rscale);
#pragma unroll
    for(int j=0;j<8;j++){
      const float* prow = PT + ((size_t)bt*DINNER + p0 + j*16 + r)*DSTATE + k;
      acc[j] = wmma_bf(cf, ldfrag_f32(prow, h), acc[j]);
    }
  }

#pragma unroll
  for(int j=0;j<8;j++){
    const int p = p0 + j*16 + r;
#pragma unroll
    for(int i=0;i<8;i++){
      const size_t m = row0 + lt*16 + i + 8*h;
      ybf[m*DINNER + p] = (__bf16)(acc[j][i] * spz[m*DINNER + p]);
    }
  }
}

// ---------------- K7: out = y @ W_out^T ----------------
// grid (ROWS/128, DMODEL/128); weight tile async double-buffered through LDS.
__global__ __launch_bounds__(256) void k_gemm2(const bf16* __restrict__ ybf,
                                               const bf16* __restrict__ w2b,
                                               float* __restrict__ out){
  __shared__ bf16 sb[2][128*32];
  const int tid = threadIdx.x;
  const int wave = tid >> 5, lane = tid & 31;
  const int h = lane >> 4, r = lane & 15;
  const int m0 = blockIdx.x*128 + wave*16;
  const int n0 = blockIdx.y*128;
  v8f acc[8] = {};
  const bf16* arow = ybf + (size_t)(m0 + r)*DINNER;

  stage_wtile(w2b, DINNER, n0, 0, sb[0], tid);
  int cur = 0;
#pragma unroll 1
  for(int k=0;k<DINNER;k+=32){
    wait_stage();
    __syncthreads();
    if(k + 32 < DINNER) stage_wtile(w2b, DINNER, n0, k + 32, sb[cur^1], tid);
    if(k + 64 < DINNER) __builtin_prefetch(arow + k + 64, 0, 0);
    const v16bf af = ldfrag(arow + k, h);
#pragma unroll
    for(int j=0;j<8;j++)
      acc[j] = wmma_bf(af, ldfrag(&sb[cur][(j*16 + r)*32], h), acc[j]);
    __syncthreads();
    cur ^= 1;
  }
#pragma unroll
  for(int j=0;j<8;j++){
    const int n = n0 + j*16 + r;
#pragma unroll
    for(int i=0;i<8;i++){
      const size_t m = (size_t)m0 + i + 8*h;
      out[m*DMODEL + n] = acc[j][i];
    }
  }
}

// ---------------- launcher ----------------
extern "C" void kernel_launch(void* const* d_in, const int* in_sizes, int n_in,
                              void* d_out, int out_size, void* d_ws, size_t ws_size,
                              hipStream_t stream){
  const float* u      = (const float*)d_in[0];
  const float* reward = (const float*)d_in[1];
  const float* w1     = (const float*)d_in[2];
  const float* b1     = (const float*)d_in[3];
  const float* cw     = (const float*)d_in[4];
  const float* cb     = (const float*)d_in[5];
  const float* bpw    = (const float*)d_in[6];
  const float* bpb    = (const float*)d_in[7];
  const float* alog   = (const float*)d_in[8];
  const float* w2     = (const float*)d_in[9];
  float* out = (float*)d_out;

  char* ws = (char*)d_ws;
  bf16*  w1b = (bf16*)(ws + OFF_W1B);
  bf16*  w2b = (bf16*)(ws + OFF_W2B);
  float* spz = (float*)(ws + OFF_SPZ);
  bf16*  xT  = (bf16*)(ws + OFF_XT);
  bf16*  Bbf = (bf16*)(ws + OFF_BBF);
  bf16*  BdT = (bf16*)(ws + OFF_BDT);
  bf16*  Cbf = (bf16*)(ws + OFF_CBF);
  bf16*  ybf = (bf16*)(ws + OFF_YBF);
  float* xBC = (float*)(ws + OFF_UNION);   // dead after k_conv
  float* PT  = (float*)(ws + OFF_UNION);   // written by k_states (aliases xBC)

  k_cvt   <<<dim3((NZXC*DMODEL + 255)/256), 256, 0, stream>>>(w1, w2, w1b, w2b);
  k_gemm1 <<<dim3(ROWS/128, NZXC/128),      256, 0, stream>>>(u, w1b, b1, spz, xBC);
  k_conv  <<<dim3(NBT, 4),                  256, 0, stream>>>(xBC, cw, cb, reward, bpw, bpb, alog,
                                                              xT, Bbf, BdT, Cbf);
  k_states<<<dim3(NBT, 4),                  256, 0, stream>>>(BdT, xT, PT);
  k_scan  <<<dim3((BATCH*16384)/256),       256, 0, stream>>>(PT, alog);
  k_y     <<<dim3(NBT, 2),                  256, 0, stream>>>(Cbf, Bbf, xT, PT, spz, alog, ybf);
  k_gemm2 <<<dim3(ROWS/128, DMODEL/128),    256, 0, stream>>>(ybf, w2b, out);
}
